// SpectralAttentionAdapter_59871844106552
// MI455X (gfx1250) — matrix operand order repaired and checked by reading, not compiled, not run on hardware
//
#include <hip/hip_runtime.h>
#include <cstdint>
#include <cstddef>

typedef __attribute__((ext_vector_type(2)))  float  v2f;
typedef __attribute__((ext_vector_type(8)))  float  v8f;
typedef __attribute__((ext_vector_type(16))) __bf16 v16bf;
typedef __attribute__((ext_vector_type(4)))  unsigned int uint32x4;
typedef __attribute__((ext_vector_type(8)))  int          int32x8;
typedef __attribute__((ext_vector_type(4)))  int          int32x4;

#define B_    32
#define CIN   4
#define H_    512
#define W_    512
#define HW    (H_*W_)
#define COUT  3
#define CRED  2
#define NPART 32          // partial-sum splits per (b,c)
#define TROWS 22          // halo rows per channel (16 + 6)
#define TCOLS 32          // staged cols  (x0-4 .. x0+27), = WMMA K
#define LPITCH 40         // LDS row pitch in ushorts (80B: TDM pad 16B/row)
#define PH    518         // padded mirror height (3 + 512 + 3)
#define PW    528         // padded mirror width  (4 + 512 + 12)
#define PPLANE (PH*PW)    // 273504 elems per channel plane
#define WSLOT (COUT*TROWS*LPITCH)   // ushorts per stage buffer (2640)

__device__ __forceinline__ unsigned short f2bf(float f) {
  unsigned int u = __float_as_uint(f);
  unsigned int r = (u + 0x7FFFu + ((u >> 16) & 1u)) >> 16;  // RNE
  return (unsigned short)r;
}

// ---------------- K1: per-(b,c) partial sums (deterministic, no atomics) ---
__global__ void __launch_bounds__(256)
k1_pool_partials(const float* __restrict__ x, float* __restrict__ partials) {
  int bc   = blockIdx.x >> 5;     // 0..127  (b*4+c)
  int part = blockIdx.x & 31;
  const float4* p4 = (const float4*)(x + (size_t)bc * HW + (size_t)part * (HW / NPART));
  float s = 0.f;
  for (int i = threadIdx.x; i < (HW / NPART) / 4; i += 256) {
    float4 v = p4[i];
    s += v.x + v.y + v.z + v.w;
  }
  __shared__ float red[256];
  red[threadIdx.x] = s;
  __syncthreads();
  for (int off = 128; off > 0; off >>= 1) {
    if (threadIdx.x < off) red[threadIdx.x] += red[threadIdx.x + off];
    __syncthreads();
  }
  if (threadIdx.x == 0) partials[bc * NPART + part] = red[0];
}

// ---------------- K2: channel-attention MLP (one lane per batch) ----------
__global__ void k2_att(const float* __restrict__ partials,
                       const float* __restrict__ w1, const float* __restrict__ b1,
                       const float* __restrict__ w2, const float* __restrict__ b2,
                       float* __restrict__ att) {
  int b = threadIdx.x;
  if (b >= B_) return;
  float pooled[CIN];
  for (int c = 0; c < CIN; ++c) {
    float s = 0.f;
    for (int p = 0; p < NPART; ++p) s += partials[(b * CIN + c) * NPART + p];
    pooled[c] = s * (1.f / (float)HW);
  }
  float h[CRED];
  for (int r = 0; r < CRED; ++r) {
    float a = b1[r];
    for (int c = 0; c < CIN; ++c) a += w1[r * CIN + c] * pooled[c];
    h[r] = fmaxf(a, 0.f);
  }
  for (int c = 0; c < CIN; ++c) {
    float a = b2[c];
    for (int r = 0; r < CRED; ++r) a += w2[c * CRED + r] * h[r];
    att[b * CIN + c] = 1.f / (1.f + __expf(-a));
  }
}

// ---------------- K3: fused (x*att) 1x1 conv via v_wmma_f32_16x16x4_f32 ---
// D[m=o, n=pix] = sum_c (conv_w[o,c]*att[b,c]) * x[b,c,pix] + bias[o]
// Writes fp32 out (for gating) AND a zero-padded bf16 mirror (TDM input).
__global__ void __launch_bounds__(256)
k3_conv1x1(const float* __restrict__ x, const float* __restrict__ att,
           const float* __restrict__ conv_w, const float* __restrict__ conv_b,
           float* __restrict__ outbuf, unsigned short* __restrict__ outbfp) {
  const int lane = threadIdx.x & 31;
  const int m    = lane & 15;                 // A row / B col
  const int k0   = (lane < 16) ? 0 : 2;       // K pair held by this lane half
  const int wid  = (int)((blockIdx.x * blockDim.x + threadIdx.x) >> 5);
  const int nw   = (int)((gridDim.x * blockDim.x) >> 5);
  const int TPB  = HW / 16;                   // 16-pixel tiles per batch image
  const int tot  = B_ * TPB;

  for (int t = wid; t < tot; t += nw) {
    int b  = t / TPB;
    int p0 = (t - b * TPB) * 16;

    // A: att-folded weights, 16x4 fp32 (rows >=3 zero)
    v2f a;
    a.x = (m < COUT) ? conv_w[m * CIN + k0]     * att[b * CIN + k0]     : 0.f;
    a.y = (m < COUT) ? conv_w[m * CIN + k0 + 1] * att[b * CIN + k0 + 1] : 0.f;

    // B: 4x16 pixels of x.  row K in {k0,k0+1}, col = lane&15
    const float* xb = x + (size_t)b * CIN * HW + p0 + m;
    v2f bm;
    bm.x = xb[(size_t)k0 * HW];
    bm.y = xb[(size_t)(k0 + 1) * HW];

    // C: bias on the 3 live rows (M = r for lanes<16)
    v8f c = {};
    if (lane < 16) { c[0] = conv_b[0]; c[1] = conv_b[1]; c[2] = conv_b[2]; }

    v8f d = __builtin_amdgcn_wmma_f32_16x16x4_f32(false, a, false, bm,
                                                  (short)0, c, false, false);

    if (lane < 16) {
      size_t off = (size_t)b * COUT * HW + p0 + lane;
      float* ob = outbuf + off;
      ob[0]              = d[0];
      ob[(size_t)HW]     = d[1];
      ob[(size_t)2 * HW] = d[2];
      int y = p0 >> 9, xcol = (p0 & 511) + lane;           // W_ == 512
      unsigned short* obf = outbfp +
          ((size_t)b * COUT * PH + (y + 3)) * PW + (xcol + 4);
      obf[0]                  = f2bf(d[0]);
      obf[(size_t)PPLANE]     = f2bf(d[1]);
      obf[(size_t)2 * PPLANE] = f2bf(d[2]);
    }
  }
}

// ---------------- K4: 7x7 spatial attention, double-buffered TDM + WMMA ---
// Per (c,ky): D[dy,dx] += A[dy,u] * B[u,dx];  A = staged rows (u: x0-4+u),
// B = Toeplitz(w): B[u,dx] = w[u-dx-1].  Every tile is staged by one
// tensor_load_to_lds from the zero-padded bf16 mirror (3D tile 32x22x3,
// LDS rows padded by TDM to an 80B pitch -> conflict-free ds_load_b128).
__global__ void __launch_bounds__(128)
k4_spatial(const unsigned short* __restrict__ outbfp,
           const float* __restrict__ outbuf, const float* __restrict__ sa_w,
           const float* __restrict__ sa_b, float* __restrict__ dst) {
  const int lane = threadIdx.x & 31;
  const int wslt = threadIdx.x >> 5;          // wave slot in block (0..3)
  const int n    = lane & 15;                 // B col (dx) / A row (dy)
  const int h16  = lane >> 4;                 // lane half

  __shared__ __align__(16) unsigned short stage[4][2][WSLOT];

  // ---- 21 tile-invariant Toeplitz B fragments (bf16 32x16) --------------
  v16bf Bm[COUT * 7];
  for (int c = 0; c < COUT; ++c) {
    for (int ky = 0; ky < 7; ++ky) {
      unsigned short bits[16];
      for (int j = 0; j < 8; ++j) {
        for (int s = 0; s < 2; ++s) {
          int K  = 2 * j + s + 8 * h16 + ((j >= 4) ? 8 : 0);
          int dd = K - n - 1;                 // kx
          float w = (dd >= 0 && dd < 7) ? sa_w[(c * 7 + ky) * 7 + dd] : 0.f;
          bits[2 * j + s] = f2bf(w);
        }
      }
      __builtin_memcpy(&Bm[c * 7 + ky], bits, 32);
    }
  }

  // ---- constant TDM descriptor words ------------------------------------
  // group1: data_size=2B, pad_enable, pad_interval=16DW(64B row), pad=4DW
  const unsigned g1w0 = (1u << 16) | (1u << 20) | (3u << 22) | (3u << 25);
  int32x8 g1 = { (int)g1w0,
                 (int)((unsigned)(PW & 0xFFFF) << 16),               // dim0 lo16
                 (int)((PW >> 16) | ((unsigned)(PH & 0xFFFF) << 16)),// dim0 hi/dim1 lo
                 (int)((PH >> 16) | (TCOLS << 16)),                  // dim1 hi/tile0
                 (int)(TROWS | (COUT << 16)),                        // tile1/tile2
                 (int)PW,                                            // dim0_stride lo32
                 (int)(((unsigned)PPLANE & 0xFFFFu) << 16),          // d0s hi/d1s lo16
                 (int)((unsigned)PPLANE >> 16) };                    // d1s [47:16]
  int32x4 g2 = { COUT, 0, 0, 0 };            // tensor_dim2
  int32x4 g3 = { 0, 0, 0, 0 };
  int32x8 g4 = { 0, 0, 0, 0, 0, 0, 0, 0 };

  unsigned ldsb[2];
  ldsb[0] = (unsigned)__builtin_amdgcn_readfirstlane(
                (int)(size_t)(void*)&stage[wslt][0][0]);
  ldsb[1] = (unsigned)__builtin_amdgcn_readfirstlane(
                (int)(size_t)(void*)&stage[wslt][1][0]);

  const float sab = sa_b[0];
  const int TX = W_ / 16, TY = H_ / 16;
  const int tilesPerB = TX * TY;
  const int totTiles  = B_ * tilesPerB;
  int wid = (int)((blockIdx.x * blockDim.x + threadIdx.x) >> 5);
  int nw  = (int)((gridDim.x * blockDim.x) >> 5);

  auto issueTDM = [&](int tile, unsigned lbase) {
    int b   = tile / tilesPerB;
    int rem = tile - b * tilesPerB;
    int ty  = rem / TX, tx = rem - ty * TX;
    // staged region origin (y0-3, x0-4) -> padded coords (y0, x0)
    const unsigned short* gt = outbfp +
        ((size_t)b * COUT * PH + (size_t)(ty * 16)) * PW + (tx * 16);
    unsigned long long ga = (unsigned long long)(size_t)gt;
    unsigned galo = (unsigned)__builtin_amdgcn_readfirstlane((int)(unsigned)ga);
    unsigned gahi = (unsigned)__builtin_amdgcn_readfirstlane(
                        (int)(unsigned)((ga >> 32) & 0x1FFFFFFull));
    uint32x4 g0 = { 1u,                    // count=1, user descriptor
                    lbase,                 // lds_addr
                    galo,                  // global_addr[31:0]
                    gahi | (2u << 30) };   // global_addr[56:32] | type=2
    __builtin_amdgcn_tensor_load_to_lds(g0, g1, g2, g3, g4, 0);
  };

  int p = 0;
  if (wid < totTiles) issueTDM(wid, ldsb[0]);

  for (int t = wid; t < totTiles; t += nw) {
    int tn = t + nw;
    if (tn < totTiles) {
      issueTDM(tn, ldsb[p ^ 1]);          // prefetch next tile's halo
      __builtin_amdgcn_s_wait_tensorcnt(1);  // older (current) DMA done
    } else {
      __builtin_amdgcn_s_wait_tensorcnt(0);
    }

    int b   = t / tilesPerB;
    int rem = t - b * tilesPerB;
    int ty  = rem / TX, tx = rem - ty * TX;
    int y0  = ty * 16, x0 = tx * 16;

    const unsigned short* sb = &stage[wslt][p][0];
    v8f d = {};
    for (int c = 0; c < COUT; ++c) {
      for (int ky = 0; ky < 7; ++ky) {
        // A fragment: row M = lane&15 at staged row (dy+ky); two b128 chunks
        const unsigned short* rowp = sb + (c * TROWS + n + ky) * LPITCH;
        uint4 u2[2];
        u2[0] = *(const uint4*)(rowp + 8 * h16);        // K 0..7  / 8..15
        u2[1] = *(const uint4*)(rowp + 16 + 8 * h16);   // K 16..23 / 24..31
        v16bf a;
        __builtin_memcpy(&a, u2, 32);
        d = __builtin_amdgcn_wmma_f32_16x16x32_bf16(false, a, false, Bm[c * 7 + ky],
                                                    (short)0, d, false, false);
      }
    }

    // ---- epilogue: sigmoid gate, write 3 channels of the 16x16 tile -----
    for (int r = 0; r < 8; ++r) {
      int dy = r + 8 * h16, dx = n;
      size_t base = (size_t)b * COUT * HW + (size_t)(y0 + dy) * W_ + (x0 + dx);
      float sAtt = 1.f / (1.f + __expf(-(d[r] + sab)));
      for (int c = 0; c < COUT; ++c) {
        float v = outbuf[base + (size_t)c * HW];
        __builtin_nontemporal_store(v * sAtt, dst + base + (size_t)c * HW);
      }
    }
    p ^= 1;
  }
}

// --------------------------------------------------------------------------
extern "C" void kernel_launch(void* const* d_in, const int* in_sizes, int n_in,
                              void* d_out, int out_size, void* d_ws, size_t ws_size,
                              hipStream_t stream) {
  const float* x      = (const float*)d_in[0];
  const float* ca_w1  = (const float*)d_in[1];
  const float* ca_b1  = (const float*)d_in[2];
  const float* ca_w2  = (const float*)d_in[3];
  const float* ca_b2  = (const float*)d_in[4];
  const float* conv_w = (const float*)d_in[5];
  const float* conv_b = (const float*)d_in[6];
  const float* sa_w   = (const float*)d_in[7];
  const float* sa_b   = (const float*)d_in[8];
  float* out = (float*)d_out;

  float* ws              = (float*)d_ws;
  float* partials        = ws;                          // 4096 floats
  float* att             = ws + 4096;                   // 128 floats
  float* outbuf          = ws + 4096 + 256;             // 25165824 floats (96 MiB)
  unsigned short* outbfp = (unsigned short*)(outbuf + (size_t)COUT * B_ * HW);
  size_t padBytes        = (size_t)B_ * COUT * PH * PW * sizeof(unsigned short);

  // zero the padded bf16 mirror so every TDM tile is in-bounds & zero-padded
  hipMemsetAsync(outbfp, 0, padBytes, stream);

  k1_pool_partials<<<B_ * CIN * NPART, 256, 0, stream>>>(x, partials);
  k2_att<<<1, 32, 0, stream>>>(partials, ca_w1, ca_b1, ca_w2, ca_b2, att);
  k3_conv1x1<<<2048, 256, 0, stream>>>(x, att, conv_w, conv_b, outbuf, outbfp);
  k4_spatial<<<2048, 128, 0, stream>>>(outbfp, outbuf, sa_w, sa_b, out);
}
